// BiLSTM_CRF_34909494182593
// MI455X (gfx1250) — matrix operand order repaired and checked by reading, not compiled, not run on hardware
//
#include <hip/hip_runtime.h>

typedef __attribute__((ext_vector_type(16))) __bf16 v16bf;
typedef __attribute__((ext_vector_type(8)))  float  v8f;
typedef __attribute__((ext_vector_type(4)))  float  v4f;   // native vector for NT stores

#define SEQ  512
#define BSZ  128
#define EMBD 128
#define HD   128      // per-direction hidden
#define G4   512      // 4*H gates
#define NTAG 7
#define TBOS 4
#define TEOS 5
#define NEGV -10000.0f

// ---------------------------------------------------------------------------
// WMMA operand packing for wave32: A and B both use contiguous k = 16*half + s
// ---------------------------------------------------------------------------

// A: 16x32 bf16 tile, src row-major (ld elems/row), top-left at src.
__device__ __forceinline__ v16bf packA(const __bf16* src, int ld) {
  const int lane = threadIdx.x & 31;
  const int m = lane & 15, half = lane >> 4;
  v16bf a;
#pragma unroll
  for (int i = 0; i < 8; ++i) {
    const int kbase = half * 16 + i * 2;
    a[2 * i]     = src[m * ld + kbase];
    a[2 * i + 1] = src[m * ld + kbase + 1];
  }
  return a;
}

// B: 32x16 tile with B[k][n] = W[n0+n][k0+k], W row-major (ldw elems/row).
__device__ __forceinline__ v16bf packB(const __bf16* W, int n0, int k0, int ldw) {
  const int lane = threadIdx.x & 31;
  const int n = lane & 15, half = lane >> 4;
  const __bf16* p = W + (size_t)(n0 + n) * ldw + k0 + half * 16;
  v16bf b;
#pragma unroll
  for (int i = 0; i < 16; ++i) b[i] = p[i];
  return b;
}

// ---------------------------------------------------------------------------
// K0: convert all four LSTM weight matrices (512x128 each) fp32 -> bf16
// ---------------------------------------------------------------------------
__global__ void k_cvt_w(const float* wf, const float* whf,
                        const float* wb, const float* whb,
                        __bf16* owf, __bf16* owhf, __bf16* owb, __bf16* owhb) {
  const int i = blockIdx.x * blockDim.x + threadIdx.x;
  if (i < G4 * EMBD) {
    owf[i]  = (__bf16)wf[i];
    owhf[i] = (__bf16)whf[i];
    owb[i]  = (__bf16)wb[i];
    owhb[i] = (__bf16)whb[i];
  }
}

// ---------------------------------------------------------------------------
// K1: embedding gather, fp32 table -> bf16 activations (S*B, 128)
// ---------------------------------------------------------------------------
__global__ void k_gather(const int* __restrict__ sen, const float* __restrict__ emb,
                         __bf16* __restrict__ xbf) {
  const size_t idx = (size_t)blockIdx.x * blockDim.x + threadIdx.x;  // < 65536*128
  const size_t sb = idx >> 7;
  const int e = (int)(idx & 127);
  const int tok = sen[sb];
  xbf[idx] = (__bf16)emb[(size_t)tok * EMBD + e];
}

// ---------------------------------------------------------------------------
// K2: input projection GEMM  xw = x(bf16) @ Wih^T + b   (M=65536,N=512,K=128)
// Output stored in WMMA-fragment-major layout:
//   xw[ ((mtile*32 + ntile)*32 + lane) * 8 + i ]  == C-fragment element i
// so the recurrence can stream it back with b128 transactions (and async DMA).
// Stored non-temporally: 268MB stream, keep it out of L2.
// ---------------------------------------------------------------------------
__global__ void __launch_bounds__(256)
k_inproj(const __bf16* __restrict__ xbf,
         const __bf16* __restrict__ wih_f, const __bf16* __restrict__ wih_b,
         const float* __restrict__ b_f, const float* __restrict__ b_b,
         float* __restrict__ xw_f, float* __restrict__ xw_b) {
  const int mtile = blockIdx.x;          // 0..4095
  const int dir   = blockIdx.y;          // 0=f, 1=b
  const __bf16* W    = dir ? wih_b : wih_f;
  const float*  bias = dir ? b_b   : b_f;
  float*        out  = dir ? xw_b  : xw_f;
  const int wave = threadIdx.x >> 5, lane = threadIdx.x & 31;
  const int n = lane & 15;
  const __bf16* Asrc = xbf + (size_t)mtile * 16 * EMBD;

  __builtin_prefetch(Asrc + 16 * EMBD, 0, 1);  // next block's A tile

  v8f acc[4] = {};
#pragma unroll
  for (int ks = 0; ks < 4; ++ks) {
    const v16bf a = packA(Asrc + ks * 32, EMBD);
#pragma unroll
    for (int j = 0; j < 4; ++j) {
      const v16bf b = packB(W, (wave * 4 + j) * 16, ks * 32, EMBD);
      acc[j] = __builtin_amdgcn_wmma_f32_16x16x32_bf16(
          false, a, false, b, (short)0, acc[j], false, false);
    }
  }
#pragma unroll
  for (int j = 0; j < 4; ++j) {
    const int ntile = wave * 4 + j;
    const float bv = bias[ntile * 16 + n];
    float* dst = out + (((size_t)mtile * 32 + ntile) * 32 + lane) * 8;
    const v4f lo = {acc[j][0] + bv, acc[j][1] + bv, acc[j][2] + bv, acc[j][3] + bv};
    const v4f hi = {acc[j][4] + bv, acc[j][5] + bv, acc[j][6] + bv, acc[j][7] + bv};
    __builtin_nontemporal_store(lo, (v4f*)dst);
    __builtin_nontemporal_store(hi, (v4f*)(dst + 4));
  }
}

// ---------------------------------------------------------------------------
// K3: LSTM recurrence. 16 blocks: dir = blk>>3, batch slice = 16*(blk&7).
// Whh B-tiles pinned in VGPRs (16 x v16bf per wave); h,c state in LDS.
// Next step's gate preactivations are prefetched into wave-private LDS with
// GLOBAL_LOAD_ASYNC_TO_LDS_B128 (ASYNCcnt), double-buffered, so the 268MB
// xw stream is hidden behind the WMMA + gate math of the current step.
// ---------------------------------------------------------------------------
__global__ void __launch_bounds__(256)
k_lstm(const float* __restrict__ xw_f, const float* __restrict__ xw_b,
       const __bf16* __restrict__ whh_f, const __bf16* __restrict__ whh_b,
       float* __restrict__ hf, float* __restrict__ hb) {
  const int dir    = blockIdx.x >> 3;
  const int bslice = blockIdx.x & 7;
  const int b0     = bslice * 16;
  const int tid    = threadIdx.x;
  const int wave   = tid >> 5, lane = tid & 31;
  const int n = lane & 15, half = lane >> 4;

  const float*  xw  = dir ? xw_b  : xw_f;
  const __bf16* whh = dir ? whh_b : whh_f;
  float*        ho  = dir ? hb    : hf;

  __shared__ float  xwbuf[2][8][4][32][8];   // 64KB: double-buffered async dest
  __shared__ __bf16 h_bf[16][HD];            // 4KB
  __shared__ float  cst[16][HD];             // 8KB
  __shared__ float  gates[16][G4];           // 32KB

  // Whh resident in registers for the whole sequence
  v16bf Bt[4][4];
#pragma unroll
  for (int j = 0; j < 4; ++j)
#pragma unroll
    for (int ks = 0; ks < 4; ++ks)
      Bt[j][ks] = packB(whh, (wave * 4 + j) * 16, ks * 32, HD);

  for (int e = tid; e < 16 * HD; e += 256) {
    (&h_bf[0][0])[e] = (__bf16)0.0f;
    (&cst[0][0])[e]  = 0.0f;
  }

  // async prefetch of one step's fragments for this wave (8 x b128 per wave)
  auto issue_prefetch = [&](int tt, int buf) {
#pragma unroll
    for (int j = 0; j < 4; ++j) {
      const int ntile = wave * 4 + j;
      const float* g =
          xw + ((((size_t)tt * 8 + bslice) * 32 + ntile) * 32 + lane) * 8;
      const unsigned ldsa = (unsigned)(uintptr_t)&xwbuf[buf][wave][j][lane][0];
      asm volatile("global_load_async_to_lds_b128 %0, %1, off"
                   :: "v"(ldsa), "v"(g) : "memory");
      asm volatile("global_load_async_to_lds_b128 %0, %1, off offset:16"
                   :: "v"(ldsa), "v"(g) : "memory");
    }
  };

  issue_prefetch(dir ? SEQ - 1 : 0, 0);
  __syncthreads();

  for (int t = 0; t < SEQ; ++t) {
    const int tt  = dir ? (SEQ - 1 - t) : t;
    const int buf = t & 1;

    if (t + 1 < SEQ) {
      issue_prefetch(dir ? (SEQ - 2 - t) : (t + 1), buf ^ 1);
      // in-order completion: all but the 8 just-issued are done => buf ready
      asm volatile("s_wait_asynccnt 0x8" ::: "memory");
    } else {
      asm volatile("s_wait_asynccnt 0x0" ::: "memory");
    }

    // C initialized with prefetched x-projection (bias folded in at K2)
    v8f acc[4];
#pragma unroll
    for (int j = 0; j < 4; ++j)
#pragma unroll
      for (int i = 0; i < 8; ++i)
        acc[j][i] = xwbuf[buf][wave][j][lane][i];

#pragma unroll
    for (int ks = 0; ks < 4; ++ks) {
      const v16bf a = packA(&h_bf[0][ks * 32], HD);   // LDS reads
#pragma unroll
      for (int j = 0; j < 4; ++j)
        acc[j] = __builtin_amdgcn_wmma_f32_16x16x32_bf16(
            false, a, false, Bt[j][ks], (short)0, acc[j], false, false);
    }
#pragma unroll
    for (int j = 0; j < 4; ++j) {
      const int n0 = (wave * 4 + j) * 16;
#pragma unroll
      for (int i = 0; i < 8; ++i)
        gates[i + 8 * half][n0 + n] = acc[j][i];
    }
    __syncthreads();

    // gate nonlinearity: 2048 elems / 256 threads
#pragma unroll
    for (int r = 0; r < 8; ++r) {
      const int e = tid + r * 256;
      const int row = e >> 7, hh = e & 127;
      const float ig = gates[row][hh];
      const float fg = gates[row][HD + hh];
      const float gg = gates[row][2 * HD + hh];
      const float og = gates[row][3 * HD + hh];
      const float si = 1.0f / (1.0f + __expf(-ig));
      const float sf = 1.0f / (1.0f + __expf(-fg));
      const float so = 1.0f / (1.0f + __expf(-og));
      const float c  = sf * cst[row][hh] + si * tanhf(gg);
      const float h  = so * tanhf(c);
      cst[row][hh]   = c;
      h_bf[row][hh]  = (__bf16)h;
      ho[((size_t)tt * BSZ + b0 + row) * HD + hh] = h;
    }
    __syncthreads();
  }
}

// ---------------------------------------------------------------------------
// K4: fts = [hf|hb] @ W_out^T + b_out   (65536 x 256) @ (256 x 7)
// ---------------------------------------------------------------------------
__global__ void __launch_bounds__(256)
k_fts(const float* __restrict__ hf, const float* __restrict__ hb,
      const float* __restrict__ W_out, const float* __restrict__ b_out,
      float* __restrict__ fts) {
  __shared__ float Wl[NTAG][256];
  __shared__ float bl[NTAG];
  for (int i = threadIdx.x; i < NTAG * 256; i += 256) (&Wl[0][0])[i] = W_out[i];
  if (threadIdx.x < NTAG) bl[threadIdx.x] = b_out[threadIdx.x];
  __syncthreads();

  const size_t sb = (size_t)blockIdx.x * 256 + threadIdx.x;
  const v4f* hfr = (const v4f*)(hf + sb * HD);
  const v4f* hbr = (const v4f*)(hb + sb * HD);
  float accv[NTAG];
#pragma unroll
  for (int tg = 0; tg < NTAG; ++tg) accv[tg] = bl[tg];
  for (int k = 0; k < HD / 4; ++k) {
    const v4f a = hfr[k], b = hbr[k];
#pragma unroll
    for (int tg = 0; tg < NTAG; ++tg) {
      const v4f wa = *(const v4f*)&Wl[tg][k * 4];
      const v4f wb = *(const v4f*)&Wl[tg][HD + k * 4];
      accv[tg] += a.x * wa.x + a.y * wa.y + a.z * wa.z + a.w * wa.w +
                  b.x * wb.x + b.y * wb.y + b.z * wb.z + b.w * wb.w;
    }
  }
#pragma unroll
  for (int tg = 0; tg < NTAG; ++tg) fts[sb * NTAG + tg] = accv[tg];
}

// ---------------------------------------------------------------------------
// K5: batched Viterbi + backtrace. 1 block, thread b handles batch lane b.
// out[0:128] = path_score, out[128 + s*128 + b] = tag (as float).
// ---------------------------------------------------------------------------
__global__ void __launch_bounds__(128)
k_viterbi(const float* __restrict__ fts, const float* __restrict__ trans,
          unsigned char* __restrict__ bptr, float* __restrict__ out) {
  const int b = threadIdx.x;
  __shared__ float tr[NTAG][NTAG];
  if (threadIdx.x < NTAG * NTAG) (&tr[0][0])[threadIdx.x] = trans[threadIdx.x];
  __syncthreads();

  float fv[NTAG];
#pragma unroll
  for (int t = 0; t < NTAG; ++t) fv[t] = (t == TBOS) ? 0.0f : NEGV;

  for (int s = 0; s < SEQ; ++s) {
    const float* ft = fts + ((size_t)s * BSZ + b) * NTAG;
    float nfv[NTAG];
#pragma unroll
    for (int nx = 0; nx < NTAG; ++nx) {
      float best = fv[0] + tr[nx][0];
      int bi = 0;
#pragma unroll
      for (int pv = 1; pv < NTAG; ++pv) {
        const float sc = fv[pv] + tr[nx][pv];
        if (sc > best) { best = sc; bi = pv; }
      }
      bptr[((size_t)s * BSZ + b) * NTAG + nx] = (unsigned char)bi;
      nfv[nx] = best + ft[nx];
    }
#pragma unroll
    for (int nx = 0; nx < NTAG; ++nx) fv[nx] = nfv[nx];
  }

  float best = fv[0] + tr[TEOS][0];
  int cur = 0;
#pragma unroll
  for (int t = 1; t < NTAG; ++t) {
    const float sc = fv[t] + tr[TEOS][t];
    if (sc > best) { best = sc; cur = t; }
  }
  out[b] = best;

  float* tag = out + BSZ;
  for (int s = SEQ - 1; s >= 0; --s) {
    tag[(size_t)s * BSZ + b] = (float)cur;
    cur = bptr[((size_t)s * BSZ + b) * NTAG + cur];
  }
}

// ---------------------------------------------------------------------------
extern "C" void kernel_launch(void* const* d_in, const int* in_sizes, int n_in,
                              void* d_out, int out_size, void* d_ws, size_t ws_size,
                              hipStream_t stream) {
  const int*   sen   = (const int*)d_in[0];
  const float* emb   = (const float*)d_in[1];
  const float* Wih_f = (const float*)d_in[2];
  const float* Whh_f = (const float*)d_in[3];
  const float* b_f   = (const float*)d_in[4];
  const float* Wih_b = (const float*)d_in[5];
  const float* Whh_b = (const float*)d_in[6];
  const float* b_b   = (const float*)d_in[7];
  const float* W_out = (const float*)d_in[8];
  const float* b_out = (const float*)d_in[9];
  const float* trans = (const float*)d_in[10];
  float* out = (float*)d_out;

  // workspace carve-up (256B aligned)
  char* p = (char*)d_ws;
  auto take = [&](size_t bytes) {
    char* r = p;
    p += (bytes + 255) & ~(size_t)255;
    return r;
  };
  const size_t M = (size_t)SEQ * BSZ;                     // 65536 rows
  __bf16* xbf     = (__bf16*)take(M * EMBD * sizeof(__bf16));
  __bf16* wihf_bf = (__bf16*)take((size_t)G4 * EMBD * sizeof(__bf16));
  __bf16* whhf_bf = (__bf16*)take((size_t)G4 * HD * sizeof(__bf16));
  __bf16* wihb_bf = (__bf16*)take((size_t)G4 * EMBD * sizeof(__bf16));
  __bf16* whhb_bf = (__bf16*)take((size_t)G4 * HD * sizeof(__bf16));
  float*  xw_f    = (float*)take(M * G4 * sizeof(float));
  float*  xw_b    = (float*)take(M * G4 * sizeof(float));
  float*  hf      = (float*)take(M * HD * sizeof(float));
  float*  hb      = (float*)take(M * HD * sizeof(float));
  float*  fts     = (float*)take(M * NTAG * sizeof(float));
  unsigned char* bptr = (unsigned char*)take(M * NTAG);

  k_cvt_w<<<(G4 * EMBD + 255) / 256, 256, 0, stream>>>(
      Wih_f, Whh_f, Wih_b, Whh_b, wihf_bf, whhf_bf, wihb_bf, whhb_bf);

  k_gather<<<(unsigned)(M * EMBD / 256), 256, 0, stream>>>(sen, emb, xbf);

  k_inproj<<<dim3(M / 16, 2), 256, 0, stream>>>(
      xbf, wihf_bf, wihb_bf, b_f, b_b, xw_f, xw_b);

  k_lstm<<<16, 256, 0, stream>>>(xw_f, xw_b, whhf_bf, whhb_bf, hf, hb);

  k_fts<<<(unsigned)(M / 256), 256, 0, stream>>>(hf, hb, W_out, b_out, fts);

  k_viterbi<<<1, 128, 0, stream>>>(fts, trans, bptr, out);
}
